// TimeSeriesMamba_35364760715924
// MI455X (gfx1250) — matrix operand order repaired and checked by reading, not compiled, not run on hardware
//
#include <hip/hip_runtime.h>
#include <hip/hip_bf16.h>

// ---------------------------------------------------------------------------
// Mamba block forward for MI455X (gfx1250, wave32, WMMA).
// GEMMs run in bf16 via V_WMMA_F32_16X16X32_BF16 (f32 accumulate); the
// sequential selective scan keeps 16 f32 states per channel in VGPRs.
// All GEMM dims are padded to tile multiples -> guard-free hot loops.
// ---------------------------------------------------------------------------

#define BATCH   2
#define SEQLEN  1024
#define DMODEL  1024
#define DINNER  2048
#define DSTATE  16
#define DTRANK  64
#define NX      (DTRANK + 2 * DSTATE)   // 96 (logical)
#define NXP     128                     // padded to tile multiple
#define MROWS   (BATCH * SEQLEN)        // 2048

typedef float v8f      __attribute__((ext_vector_type(8)));
typedef __bf16 bf16x16 __attribute__((ext_vector_type(16)));
typedef unsigned short u16x8 __attribute__((ext_vector_type(8)));

union ABFrag { bf16x16 v; u16x8 h[2]; };

__device__ __forceinline__ unsigned short f2bf(float f) {
  unsigned int u = __float_as_uint(f);
  unsigned int r = u + 0x7FFFu + ((u >> 16) & 1u);   // round-to-nearest-even
  return (unsigned short)(r >> 16);
}

__device__ __forceinline__ float silu(float v) {
  return v / (1.f + __expf(-v));
}

#define WMMA_BF16(a, b, c) \
  __builtin_amdgcn_wmma_f32_16x16x32_bf16(false, (a), false, (b), (short)0, (c), false, false)

// ---------------------------------------------------------------------------
// bf16 GEMM:  C[M,N] (f32, ldc) = A[M,K] (bf16) * B[N,K]^T (bf16)
// Block tile 64x128, BK=64, 256 threads = 8 waves in a 2x4 wave grid.
// Each wave computes a 32x32 slab: 2x2 register tiles -> 8 WMMA per K-tile,
// 2 ds_load_b128 per WMMA.  Global staging pointers are strength-reduced.
// Requires: M mult of 64, N mult of 128, K mult of 64.  No bounds checks.
// ---------------------------------------------------------------------------
__global__ __launch_bounds__(256) void gemm_bf16_wmma(
    const unsigned short* __restrict__ A, const unsigned short* __restrict__ Bw,
    float* __restrict__ C, int M, int N, int K, int ldc) {
  __shared__ __align__(16) unsigned short Alds[64][72];   // 64 halves + 8 pad
  __shared__ __align__(16) unsigned short Blds[128][72];

  const int tid  = threadIdx.x;
  const int lane = tid & 31;
  const int wave = tid >> 5;       // 0..7
  const int wm   = wave >> 2;      // 0..1  (M direction)
  const int wn   = wave & 3;       // 0..3  (N direction)
  const int kh   = lane >> 4;      // lane half (0/1)
  const int l16  = lane & 15;

  const int m0 = blockIdx.y * 64;
  const int n0 = blockIdx.x * 128;

  // tile staging: 32 rows x 8 chunks of 16B; A: 2 rows/thread, B: 4 rows/thread
  const int rS = tid >> 3;         // 0..31
  const int c8 = tid & 7;          // 16B chunk within the 128B row

  const unsigned short* pA0 = A + (m0 + rS) * K + c8 * 8;
  const unsigned short* pA1 = pA0 + 32 * K;
  const unsigned short* pB0 = Bw + (n0 + rS) * K + c8 * 8;
  const unsigned short* pB1 = pB0 + 32 * K;
  const unsigned short* pB2 = pB0 + 64 * K;
  const unsigned short* pB3 = pB0 + 96 * K;

  unsigned short* sA0 = &Alds[rS][c8 * 8];
  unsigned short* sA1 = &Alds[rS + 32][c8 * 8];
  unsigned short* sB0 = &Blds[rS][c8 * 8];
  unsigned short* sB1 = &Blds[rS + 32][c8 * 8];
  unsigned short* sB2 = &Blds[rS + 64][c8 * 8];
  unsigned short* sB3 = &Blds[rS + 96][c8 * 8];

  // per-wave fragment base rows in LDS
  const unsigned short* fA0 = &Alds[wm * 32 + l16][0];
  const unsigned short* fA1 = &Alds[wm * 32 + 16 + l16][0];
  const unsigned short* fB0 = &Blds[wn * 32 + l16][0];
  const unsigned short* fB1 = &Blds[wn * 32 + 16 + l16][0];

  v8f acc00 = {}, acc01 = {}, acc10 = {}, acc11 = {};

  const int nIter = K >> 6;
  for (int it = 0; it < nIter; ++it) {
    if (it + 1 < nIter) {   // CDNA5 global_prefetch_b8 for the next K tile
      __builtin_prefetch(pA0 + 64, 0, 1);
      __builtin_prefetch(pB0 + 64, 0, 1);
      __builtin_prefetch(pB2 + 64, 0, 1);
    }
    *(u16x8*)sA0 = *(const u16x8*)pA0;
    *(u16x8*)sA1 = *(const u16x8*)pA1;
    *(u16x8*)sB0 = *(const u16x8*)pB0;
    *(u16x8*)sB1 = *(const u16x8*)pB1;
    *(u16x8*)sB2 = *(const u16x8*)pB2;
    *(u16x8*)sB3 = *(const u16x8*)pB3;
    pA0 += 64; pA1 += 64; pB0 += 64; pB1 += 64; pB2 += 64; pB3 += 64;
    __syncthreads();

#pragma unroll
    for (int ks = 0; ks < 2; ++ks) {    // two K=32 WMMA steps per tile
      // A frags (16x32): lane half selects K {0..7,16..23} / {8..15,24..31}
      ABFrag a0, a1, b0, b1;
      a0.h[0] = *(const u16x8*)&fA0[ks * 32 + kh * 8];
      a0.h[1] = *(const u16x8*)&fA0[ks * 32 + 16 + kh * 8];
      a1.h[0] = *(const u16x8*)&fA1[ks * 32 + kh * 8];
      a1.h[1] = *(const u16x8*)&fA1[ks * 32 + 16 + kh * 8];
      // B frags (32x16): lane half selects K {0..7,16..23} / {8..15,24..31}
      b0.h[0] = *(const u16x8*)&fB0[ks * 32 + kh * 8];
      b0.h[1] = *(const u16x8*)&fB0[ks * 32 + 16 + kh * 8];
      b1.h[0] = *(const u16x8*)&fB1[ks * 32 + kh * 8];
      b1.h[1] = *(const u16x8*)&fB1[ks * 32 + 16 + kh * 8];

      acc00 = WMMA_BF16(a0.v, b0.v, acc00);
      acc01 = WMMA_BF16(a0.v, b1.v, acc01);
      acc10 = WMMA_BF16(a1.v, b0.v, acc10);
      acc11 = WMMA_BF16(a1.v, b1.v, acc11);
    }
    __syncthreads();
  }

  // D layout: VGPR r -> (M = kh*8 + r, N = l16)
  const int nn = n0 + wn * 32 + l16;
#pragma unroll
  for (int r = 0; r < 8; ++r) {
    const int m = m0 + wm * 32 + kh * 8 + r;
    C[m * ldc + nn]             = acc00[r];
    C[m * ldc + nn + 16]        = acc01[r];
    C[(m + 16) * ldc + nn]      = acc10[r];
    C[(m + 16) * ldc + nn + 16] = acc11[r];
  }
}

// ---------------------------------------------------------------------------
// f32 -> bf16 conversion
// ---------------------------------------------------------------------------
__global__ __launch_bounds__(256) void f32_to_bf16_k(
    const float* __restrict__ src, unsigned short* __restrict__ dst, int n) {
  int i = blockIdx.x * 256 + threadIdx.x;
  if (i < n) dst[i] = f2bf(src[i]);
}

// W_x (96 x 2048) -> bf16 padded to 128 x 2048 (zero rows 96..127)
__global__ __launch_bounds__(256) void convert_pad_wx_k(
    const float* __restrict__ W_x, unsigned short* __restrict__ dst) {
  int i = blockIdx.x * 256 + threadIdx.x;
  if (i >= NXP * DINNER) return;
  int row = i >> 11;   // / DINNER
  dst[i] = (row < NX) ? f2bf(W_x[i]) : (unsigned short)0;
}

// dbc[:, :64] -> bf16 (dt-rank slice for the dt GEMM); dbc has ldc = NXP
__global__ __launch_bounds__(256) void slice_dt_bf16_k(
    const float* __restrict__ dbc, unsigned short* __restrict__ dst) {
  int i = blockIdx.x * 256 + threadIdx.x;
  if (i < MROWS * DTRANK) {
    int m = i >> 6, r = i & 63;
    dst[i] = f2bf(dbc[m * NXP + r]);
  }
}

// ---------------------------------------------------------------------------
// Causal depthwise conv (K=4) + bias + SiLU over xz[..., :DINNER].
// Writes f32 (for scan/skip) and bf16 (for the dbc GEMM).
// ---------------------------------------------------------------------------
__global__ __launch_bounds__(256) void conv_silu_k(
    const float* __restrict__ xz, const float* __restrict__ cw,
    const float* __restrict__ cb, float* __restrict__ xcf,
    unsigned short* __restrict__ xcb) {
  int idx = blockIdx.x * 256 + threadIdx.x;
  if (idx >= BATCH * SEQLEN * DINNER) return;
  int d = idx & (DINNER - 1);
  int l = (idx >> 11) & (SEQLEN - 1);
  int b = idx >> 21;
  float acc = cb[d];
#pragma unroll
  for (int j = 0; j < 4; ++j) {
    int lj = l - 3 + j;
    if (lj >= 0) acc += cw[d * 4 + j] * xz[((b << 10) + lj) * (2 * DINNER) + d];
  }
  float s = silu(acc);
  xcf[idx] = s;
  xcb[idx] = f2bf(s);
}

// ---------------------------------------------------------------------------
// Selective scan: one thread per (batch, channel), 16 f32 states in VGPRs.
// B/C rows (from dbc) staged in LDS 32 timesteps at a time (broadcast reads).
// Fuses softplus(dt), D-skip and the SiLU(z) gate; emits y in bf16.
// ---------------------------------------------------------------------------
__global__ __launch_bounds__(256) void mamba_scan_k(
    const float* __restrict__ dtlin, const float* __restrict__ b_dt,
    const float* __restrict__ A_log, const float* __restrict__ dbc,
    const float* __restrict__ xconv, const float* __restrict__ xz,
    const float* __restrict__ D_skip, unsigned short* __restrict__ y_bf) {
  __shared__ float bcs[32][32];    // [timestep][B(0..15) | C(16..31)]
  const int tid = threadIdx.x;
  const int b = blockIdx.y;
  const int d = blockIdx.x * 256 + tid;

  float h[DSTATE], Ad[DSTATE];
#pragma unroll
  for (int n = 0; n < DSTATE; ++n) {
    h[n] = 0.f;
    Ad[n] = -__expf(A_log[d * DSTATE + n]);
  }
  const float bdt = b_dt[d];
  const float Dsk = D_skip[d];

  for (int l0 = 0; l0 < SEQLEN; l0 += 32) {
#pragma unroll
    for (int rep = 0; rep < 4; ++rep) {
      int linear = rep * 256 + tid;      // 0..1023
      int t = linear >> 5;
      int c = linear & 31;
      bcs[t][c] = dbc[(b * SEQLEN + l0 + t) * NXP + DTRANK + c];
    }
    __syncthreads();

    for (int t = 0; t < 32; ++t) {
      const int l = l0 + t;
      const int idx = (b * SEQLEN + l) * DINNER + d;
      float dv = dtlin[idx] + bdt;
      dv = dv > 20.f ? dv : log1pf(__expf(dv));      // softplus
      const float xv = xconv[idx];
      const float xdt = xv * dv;
      float y = 0.f;
#pragma unroll
      for (int n = 0; n < DSTATE; ++n) {
        float ab = __expf(dv * Ad[n]);               // A_bar
        h[n] = ab * h[n] + xdt * bcs[t][n];
        y += bcs[t][16 + n] * h[n];
      }
      const float zv = xz[(b * SEQLEN + l) * (2 * DINNER) + DINNER + d];
      const float out = (y + xv * Dsk) * silu(zv);
      y_bf[idx] = f2bf(out);
    }
    __syncthreads();
  }
}

// ---------------------------------------------------------------------------
// Host launcher
// ---------------------------------------------------------------------------
extern "C" void kernel_launch(void* const* d_in, const int* in_sizes, int n_in,
                              void* d_out, int out_size, void* d_ws, size_t ws_size,
                              hipStream_t stream) {
  (void)in_sizes; (void)n_in; (void)out_size; (void)ws_size;
  const float* x      = (const float*)d_in[0];
  const float* W_in   = (const float*)d_in[1];
  const float* conv_w = (const float*)d_in[2];
  const float* conv_b = (const float*)d_in[3];
  const float* W_x    = (const float*)d_in[4];
  const float* W_dt   = (const float*)d_in[5];
  const float* b_dt   = (const float*)d_in[6];
  const float* A_log  = (const float*)d_in[7];
  const float* D_skip = (const float*)d_in[8];
  const float* W_out  = (const float*)d_in[9];
  float* out = (float*)d_out;

  char* ws = (char*)d_ws;
  size_t off = 0;
  auto alloc = [&](size_t bytes) -> void* {
    void* p = ws + off;
    off += (bytes + 255) & ~(size_t)255;
    return p;
  };

  unsigned short* x_bf     = (unsigned short*)alloc((size_t)MROWS * DMODEL * 2);
  unsigned short* Win_bf   = (unsigned short*)alloc((size_t)2 * DINNER * DMODEL * 2);
  unsigned short* Wx_bf    = (unsigned short*)alloc((size_t)NXP * DINNER * 2);
  unsigned short* Wdt_bf   = (unsigned short*)alloc((size_t)DINNER * DTRANK * 2);
  unsigned short* Wout_bf  = (unsigned short*)alloc((size_t)DMODEL * DINNER * 2);
  float*          xz       = (float*)alloc((size_t)MROWS * 2 * DINNER * 4);
  float*          xconv_f  = (float*)alloc((size_t)MROWS * DINNER * 4);
  unsigned short* xconv_bf = (unsigned short*)alloc((size_t)MROWS * DINNER * 2);
  float*          dbc      = (float*)alloc((size_t)MROWS * NXP * 4);
  unsigned short* dbcdt_bf = (unsigned short*)alloc((size_t)MROWS * DTRANK * 2);
  float*          dtlin    = (float*)alloc((size_t)MROWS * DINNER * 4);
  unsigned short* y_bf     = (unsigned short*)alloc((size_t)MROWS * DINNER * 2);

  // 1) convert activations + weights to bf16 (W_x zero-padded to 128 rows)
  f32_to_bf16_k<<<(MROWS * DMODEL + 255) / 256, 256, 0, stream>>>(x, x_bf, MROWS * DMODEL);
  f32_to_bf16_k<<<(2 * DINNER * DMODEL + 255) / 256, 256, 0, stream>>>(W_in, Win_bf, 2 * DINNER * DMODEL);
  convert_pad_wx_k<<<(NXP * DINNER + 255) / 256, 256, 0, stream>>>(W_x, Wx_bf);
  f32_to_bf16_k<<<(DINNER * DTRANK + 255) / 256, 256, 0, stream>>>(W_dt, Wdt_bf, DINNER * DTRANK);
  f32_to_bf16_k<<<(DMODEL * DINNER + 255) / 256, 256, 0, stream>>>(W_out, Wout_bf, DMODEL * DINNER);

  // 2) xz = x @ W_in^T   (2048 x 4096 x 1024)
  gemm_bf16_wmma<<<dim3(2 * DINNER / 128, MROWS / 64), 256, 0, stream>>>(
      x_bf, Win_bf, xz, MROWS, 2 * DINNER, DMODEL, 2 * DINNER);

  // 3) causal depthwise conv + SiLU on the x_ssm half
  conv_silu_k<<<(BATCH * SEQLEN * DINNER + 255) / 256, 256, 0, stream>>>(
      xz, conv_w, conv_b, xconv_f, xconv_bf);

  // 4) dbc = x_conv @ W_x^T   (2048 x 128(pad) x 2048), ldc = 128
  gemm_bf16_wmma<<<dim3(NXP / 128, MROWS / 64), 256, 0, stream>>>(
      xconv_bf, Wx_bf, dbc, MROWS, NXP, DINNER, NXP);

  // 5) dt_lin = dbc[:, :64] @ W_dt^T   (2048 x 2048 x 64)
  slice_dt_bf16_k<<<(MROWS * DTRANK + 255) / 256, 256, 0, stream>>>(dbc, dbcdt_bf);
  gemm_bf16_wmma<<<dim3(DINNER / 128, MROWS / 64), 256, 0, stream>>>(
      dbcdt_bf, Wdt_bf, dtlin, MROWS, DINNER, DTRANK, DINNER);

  // 6) selective scan + gate -> y (bf16)
  mamba_scan_k<<<dim3(DINNER / 256, BATCH), 256, 0, stream>>>(
      dtlin, b_dt, A_log, dbc, xconv_f, xz, D_skip, y_bf);

  // 7) out = y @ W_out^T   (2048 x 1024 x 2048)
  gemm_bf16_wmma<<<dim3(DMODEL / 128, MROWS / 64), 256, 0, stream>>>(
      y_bf, Wout_bf, out, MROWS, DMODEL, DINNER, DMODEL);
}
